// FramePrimerEncoder_39659728011315
// MI455X (gfx1250) — matrix operand order repaired and checked by reading, not compiled, not run on hardware
//
#include <hip/hip_runtime.h>

// ---------------------------------------------------------------------------
// FramePrimerEncoder for MI455X (gfx1250): all GEMMs via v_wmma_f32_16x16x32_f16
// with register-staged LDS double buffering (b128 staging) + global_prefetch_b8.
// All GEMM operands are staged via the NT path (contiguous b128); transposed
// operands (rel_w^T, v^T) are materialized once by tiled transpose kernels.
// WMMA kernels run 1 workgroup (8 wave32) per WGP: amdgpu_waves_per_eu(2)
// lifts the default VGPR budget so staged tiles never spill.
// B=4, T(w)=1024, C(h=BINS)=1024, BANDS=4, D=256, F=4096
// d_out floats: [0,4M)=out, [4M,20M)=qk (pre-softmax, also softmax input).
// ---------------------------------------------------------------------------

typedef __attribute__((ext_vector_type(16))) _Float16 v16h;
typedef __attribute__((ext_vector_type(8)))  float    v8f;

union F16x16 { v16h v; uint4 q[2]; _Float16 h[16]; };

#define TILE  128
#define KSTEP 32

// ---- WMMA fragment loads from LDS (layouts per CDNA5 ISA 7.12.2, wave32) ---
__device__ __forceinline__ v16h ld_afrag(const _Float16* sm, int fm, int lane) {
  int hh = lane >> 4, m = lane & 15;
  const _Float16* base = sm + (fm * 16 + m) * KSTEP + 8 * hh;
  F16x16 f;
  f.q[0] = *(const uint4*)(base);
  f.q[1] = *(const uint4*)(base + 16);
  return f.v;
}
__device__ __forceinline__ v16h ld_bfrag(const _Float16* sm, int fn, int lane) {
  int hh = lane >> 4, n = lane & 15;
  const _Float16* base = sm + (fn * 16 + n) * KSTEP + 8 * hh;
  F16x16 f;
  f.q[0] = *(const uint4*)(base);
  f.q[1] = *(const uint4*)(base + 16);
  return f.v;
}

// ---- Register-staged tile loaders. LD is a compile-time leading dim so all
// per-iteration offsets become instruction immediates. Row-major source
// [row][LD]; stages rows row0..+127, cols k0..+31 into LDS [128][32].
// Thread (tid>>1, tid&1) owns a contiguous 16-f16 half row. ------------------
template <int LD>
struct StageNT_H {                 // f16 source: 2x b128 loads, 2x ds_store_b128
  const _Float16* src;
  int voff;
  uint4 r4[2];
  __device__ __forceinline__ void init(const void* s, long row0, int tid) {
    src = (const _Float16*)s + row0 * LD;
    voff = (tid >> 1) * LD + (tid & 1) * 16;
  }
  __device__ __forceinline__ void load(int k0) {
    const _Float16* p = src + voff + k0;
    r4[0] = *(const uint4*)(p);
    r4[1] = *(const uint4*)(p + 8);
  }
  __device__ __forceinline__ void prefetch(int k0) const {
    __builtin_prefetch(src + voff + k0, 0, 1);
  }
  __device__ __forceinline__ void store(_Float16* dst, int tid) {
    _Float16* d = dst + (tid >> 1) * KSTEP + (tid & 1) * 16;
    *(uint4*)(d)     = r4[0];
    *(uint4*)(d + 8) = r4[1];
  }
};
template <int LD>
struct StageNT_F {                 // f32 source (weights): 4x b128 loads + cvt
  const float* src;
  int voff;
  float4 rf[4];
  __device__ __forceinline__ void init(const void* s, long row0, int tid) {
    src = (const float*)s + row0 * LD;
    voff = (tid >> 1) * LD + (tid & 1) * 16;
  }
  __device__ __forceinline__ void load(int k0) {
    const float* p = src + voff + k0;
    #pragma unroll
    for (int i = 0; i < 4; ++i) rf[i] = *(const float4*)(p + i * 4);
  }
  __device__ __forceinline__ void prefetch(int k0) const {
    __builtin_prefetch(src + voff + k0, 0, 1);
  }
  __device__ __forceinline__ void store(_Float16* dst, int tid) {
    _Float16* d = dst + (tid >> 1) * KSTEP + (tid & 1) * 16;
    F16x16 t;
    #pragma unroll
    for (int i = 0; i < 4; ++i) {
      t.h[i * 4 + 0] = (_Float16)rf[i].x;
      t.h[i * 4 + 1] = (_Float16)rf[i].y;
      t.h[i * 4 + 2] = (_Float16)rf[i].z;
      t.h[i * 4 + 3] = (_Float16)rf[i].w;
    }
    *(uint4*)(d)     = t.q[0];
    *(uint4*)(d + 8) = t.q[1];
  }
};

// ---- One K-step: publish regs -> LDS, start next fetch, run 8 WMMAs.
// bf[2] cached (16 VGPRs); af loaded per-fm (8 live) to limit pressure. -----
template <class ASt, class BSt>
__device__ __forceinline__ void gemm_step(ASt& as, BSt& bs, v8f (&acc)[4][2],
                                          _Float16* bufA, _Float16* bufB,
                                          int k0, int K, int tid, int lane,
                                          int wm, int wn) {
  __syncthreads();
  as.store(bufA, tid);
  bs.store(bufB, tid);
  __syncthreads();
  if (k0 + KSTEP < K)     { as.load(k0 + KSTEP); bs.load(k0 + KSTEP); }
  if (k0 + 2 * KSTEP < K) { as.prefetch(k0 + 2 * KSTEP); bs.prefetch(k0 + 2 * KSTEP); }
  v16h bf[2];
  #pragma unroll
  for (int fn = 0; fn < 2; ++fn) bf[fn] = ld_bfrag(bufB + wn * 32 * KSTEP, fn, lane);
  #pragma unroll
  for (int fm = 0; fm < 4; ++fm) {
    v16h af = ld_afrag(bufA + wm * 64 * KSTEP, fm, lane);
    #pragma unroll
    for (int fn = 0; fn < 2; ++fn)
      acc[fm][fn] = __builtin_amdgcn_wmma_f32_16x16x32_f16(
          false, af, false, bf[fn], (short)0, acc[fm][fn], false, false);
  }
}

template <class ASt, class BSt>
__device__ __forceinline__ void gemm_loop(ASt& as, BSt& bs, v8f (&acc)[4][2],
                                          _Float16* smA0, _Float16* smB0,
                                          _Float16* smA1, _Float16* smB1,
                                          int K, int tid, int lane, int wm, int wn) {
  as.load(0); bs.load(0);
  #pragma unroll 1
  for (int k0 = 0; k0 < K; k0 += 2 * KSTEP) {
    gemm_step(as, bs, acc, smA0, smB0, k0,         K, tid, lane, wm, wn);
    gemm_step(as, bs, acc, smA1, smB1, k0 + KSTEP, K, tid, lane, wm, wn);
  }
}

// EPI: 0 = f16 store, 1 = relu^2 f16, 2 = f32 residual add, 3 = f32 * 1/32
template <int EPI, int LDC>
__device__ __forceinline__ void epilogue(v8f (&acc)[4][2], void* C,
                                         const float* Res, long tm, long tn,
                                         int lane, int wm, int wn) {
  int hh = lane >> 4, n16 = lane & 15;
  #pragma unroll
  for (int fm = 0; fm < 4; ++fm)
    #pragma unroll
    for (int fn = 0; fn < 2; ++fn)
      #pragma unroll
      for (int r = 0; r < 8; ++r) {
        long row = tm + wm * 64 + fm * 16 + r + 8 * hh;
        long col = tn + wn * 32 + fn * 16 + n16;
        float val = acc[fm][fn][r];
        if (EPI == 0) {
          ((_Float16*)C)[row * LDC + col] = (_Float16)val;
        } else if (EPI == 1) {
          float rl = fmaxf(val, 0.0f);
          ((_Float16*)C)[row * LDC + col] = (_Float16)(rl * rl);
        } else if (EPI == 2) {
          ((float*)C)[row * LDC + col] = Res[row * LDC + col] + val;
        } else {
          ((float*)C)[row * LDC + col] = val * 0.03125f;  // 1/sqrt(1024)
        }
      }
}

// ---------------------------------------------------------------------------
// Generic 128x128 WMMA GEMM, compile-time dims. One workgroup (8 waves) per
// WGP = 2 waves/SIMD32: declare that so the allocator's VGPR budget covers the
// staged tiles without spilling.
// ---------------------------------------------------------------------------
template <class ASt, class BSt, int EPI, int LDC, int K>
__global__ __launch_bounds__(256)
__attribute__((amdgpu_waves_per_eu(2)))
void k_gemm(const void* __restrict__ A, const void* __restrict__ B,
            void* __restrict__ C, const float* __restrict__ Res) {
  __shared__ __align__(16) _Float16 smA[2][TILE * KSTEP];
  __shared__ __align__(16) _Float16 smB[2][TILE * KSTEP];
  int tid = threadIdx.x;
  int lane = tid & 31, wave = tid >> 5;
  int wm = wave & 1, wn = wave >> 1;                  // 2x4 waves -> 64x32 each
  long tm = (long)blockIdx.y * TILE;
  long tn = (long)blockIdx.x * TILE;

  v8f acc[4][2];
  #pragma unroll
  for (int i = 0; i < 4; ++i)
    #pragma unroll
    for (int j = 0; j < 2; ++j)
      acc[i][j] = (v8f){0.f, 0.f, 0.f, 0.f, 0.f, 0.f, 0.f, 0.f};

  ASt as; BSt bs;
  as.init(A, tm, tid);
  bs.init(B, tn, tid);
  gemm_loop(as, bs, acc, smA[0], smB[0], smA[1], smB[1], K, tid, lane, wm, wn);
  epilogue<EPI, LDC>(acc, C, Res, tm, tn, lane, wm, wn);
}

// ---------------------------------------------------------------------------
// Attention scores (one band, one batch): two accumulation passes,
//   qk[t1,t2] = ( q[t1]·k[t2] + relT[t1]·q[t2] ) / 32  -> f32 into d_out
// relT is the pre-transposed rel_w (f16 [1024][256]).
// ---------------------------------------------------------------------------
__global__ __launch_bounds__(256)
__attribute__((amdgpu_waves_per_eu(2)))
void k_scores(const _Float16* __restrict__ qh, const _Float16* __restrict__ kh,
              const _Float16* __restrict__ relT, float* __restrict__ qkOut) {
  __shared__ __align__(16) _Float16 smA[2][TILE * KSTEP];
  __shared__ __align__(16) _Float16 smB[2][TILE * KSTEP];
  int tid = threadIdx.x;
  int lane = tid & 31, wave = tid >> 5;
  int wm = wave & 1, wn = wave >> 1;
  long tm = (long)blockIdx.y * TILE;   // t1
  long tn = (long)blockIdx.x * TILE;   // t2

  v8f acc[4][2];
  #pragma unroll
  for (int i = 0; i < 4; ++i)
    #pragma unroll
    for (int j = 0; j < 2; ++j)
      acc[i][j] = (v8f){0.f, 0.f, 0.f, 0.f, 0.f, 0.f, 0.f, 0.f};

  {  // pass 1: q[t1] . k[t2] over d=256
    StageNT_H<1024> as; as.init(qh, tm, tid);
    StageNT_H<1024> bs; bs.init(kh, tn, tid);
    gemm_loop(as, bs, acc, smA[0], smB[0], smA[1], smB[1], 256, tid, lane, wm, wn);
  }
  {  // pass 2: relT[t1] . q[t2] over d=256 (both NT staged)
    StageNT_H<256>  as; as.init(relT, tm, tid);
    StageNT_H<1024> bs; bs.init(qh, tn, tid);
    gemm_loop(as, bs, acc, smA[0], smB[0], smA[1], smB[1], 256, tid, lane, wm, wn);
  }
  epilogue<3, 1024>(acc, qkOut, nullptr, tm, tn, lane, wm, wn);
}

// ---- in-projection + transpose: xf[b][t][c] = w0*x[b,0,c,t]+w1*x[b,1,c,t]+bias
__global__ void k_inproj(const float* __restrict__ x, const float* __restrict__ in_w,
                         const float* __restrict__ in_b, float* __restrict__ xf) {
  __shared__ float tile[32][33];
  int b = blockIdx.z;
  int t0 = blockIdx.x * 32, c0 = blockIdx.y * 32;
  float w0 = in_w[0], w1 = in_w[1], bias = in_b[0];
  const float* x0 = x + ((long)b * 2) * 1048576;
  const float* x1 = x0 + 1048576;
  int tx = threadIdx.x, ty = threadIdx.y;
  #pragma unroll
  for (int i = 0; i < 4; ++i) {
    int c = c0 + ty + i * 8;
    long off = (long)c * 1024 + t0 + tx;
    tile[ty + i * 8][tx] = w0 * x0[off] + w1 * x1[off] + bias;
  }
  __syncthreads();
  #pragma unroll
  for (int i = 0; i < 4; ++i) {
    int t = t0 + ty + i * 8;
    xf[(long)b * 1048576 + (long)t * 1024 + c0 + tx] = tile[tx][ty + i * 8];
  }
}

// ---- tiled f16 transpose per batch: out[b][c][t] = in[b][t][c] -------------
__global__ void k_transpose_h(const _Float16* __restrict__ in,
                              _Float16* __restrict__ out) {
  __shared__ _Float16 tile[32][33];
  int b = blockIdx.z;
  int t0 = blockIdx.x * 32, c0 = blockIdx.y * 32;
  const _Float16* pi = in + (long)b * 1048576;
  _Float16* po = out + (long)b * 1048576;
  int tx = threadIdx.x, ty = threadIdx.y;
  #pragma unroll
  for (int i = 0; i < 4; ++i)
    tile[ty + i * 8][tx] = pi[(long)(t0 + ty + i * 8) * 1024 + c0 + tx];
  __syncthreads();
  #pragma unroll
  for (int i = 0; i < 4; ++i)
    po[(long)(c0 + ty + i * 8) * 1024 + t0 + tx] = tile[tx][ty + i * 8];
}

// ---- rel_w transpose: relT[t][d] = (f16) rel_w[d][t]  ([256][1024] f32 in) --
__global__ void k_transpose_rel(const float* __restrict__ in,
                                _Float16* __restrict__ out) {
  __shared__ float tile[32][33];
  int t0 = blockIdx.x * 32, d0 = blockIdx.y * 32;
  int tx = threadIdx.x, ty = threadIdx.y;
  #pragma unroll
  for (int i = 0; i < 4; ++i)
    tile[ty + i * 8][tx] = in[(long)(d0 + ty + i * 8) * 1024 + t0 + tx];
  __syncthreads();
  #pragma unroll
  for (int i = 0; i < 4; ++i)
    out[(long)(t0 + ty + i * 8) * 256 + d0 + tx] = (_Float16)tile[tx][ty + i * 8];
}

// ---- LayerNorm over last dim (1024), f32 in -> f16 out ---------------------
__global__ __launch_bounds__(256) void k_layernorm(
    const float* __restrict__ in, const float* __restrict__ g,
    const float* __restrict__ be, _Float16* __restrict__ out) {
  __shared__ float red[256];
  long row = blockIdx.x;
  const float* r = in + row * 1024;
  int tid = threadIdx.x;
  float v4[4];
  float s = 0.0f;
  #pragma unroll
  for (int i = 0; i < 4; ++i) { v4[i] = r[tid + i * 256]; s += v4[i]; }
  red[tid] = s; __syncthreads();
  for (int st = 128; st > 0; st >>= 1) { if (tid < st) red[tid] += red[tid + st]; __syncthreads(); }
  float mean = red[0] * (1.0f / 1024.0f);
  __syncthreads();
  float vs = 0.0f;
  #pragma unroll
  for (int i = 0; i < 4; ++i) { float d = v4[i] - mean; vs += d * d; }
  red[tid] = vs; __syncthreads();
  for (int st = 128; st > 0; st >>= 1) { if (tid < st) red[tid] += red[tid + st]; __syncthreads(); }
  float rstd = rsqrtf(red[0] * (1.0f / 1024.0f) + 1e-5f);
  #pragma unroll
  for (int i = 0; i < 4; ++i) {
    int c = tid + i * 256;
    out[row * 1024 + c] = (_Float16)((v4[i] - mean) * rstd * g[c] + be[c]);
  }
}

// ---- depthwise conv-3 along t with zero pad, per-batch boundaries ----------
__global__ void k_dwconv(const _Float16* __restrict__ in, const float* __restrict__ w,
                         _Float16* __restrict__ out) {
  long idx = (long)blockIdx.x * blockDim.x + threadIdx.x;  // b*T*C
  int c = (int)(idx & 1023);
  long row = idx >> 10;
  int t = (int)(row & 1023);
  float vm = (t > 0)    ? (float)in[idx - 1024] : 0.0f;
  float v0 = (float)in[idx];
  float vp = (t < 1023) ? (float)in[idx + 1024] : 0.0f;
  out[idx] = (_Float16)(w[c * 3] * vm + w[c * 3 + 1] * v0 + w[c * 3 + 2] * vp);
}

// ---- row softmax: qk f32 [rows][1024] -> prob f16 --------------------------
__global__ __launch_bounds__(256) void k_softmax(const float* __restrict__ qk,
                                                 _Float16* __restrict__ p) {
  __shared__ float red[256];
  long row = blockIdx.x;
  const float* r = qk + row * 1024;
  int tid = threadIdx.x;
  float v[4];
  float mx = -3.402823466e+38f;
  #pragma unroll
  for (int i = 0; i < 4; ++i) { v[i] = r[tid + i * 256]; mx = fmaxf(mx, v[i]); }
  red[tid] = mx; __syncthreads();
  for (int st = 128; st > 0; st >>= 1) { if (tid < st) red[tid] = fmaxf(red[tid], red[tid + st]); __syncthreads(); }
  mx = red[0];
  __syncthreads();
  float s = 0.0f;
  #pragma unroll
  for (int i = 0; i < 4; ++i) { v[i] = __expf(v[i] - mx); s += v[i]; }
  red[tid] = s; __syncthreads();
  for (int st = 128; st > 0; st >>= 1) { if (tid < st) red[tid] += red[tid + st]; __syncthreads(); }
  float inv = 1.0f / red[0];
  #pragma unroll
  for (int i = 0; i < 4; ++i)
    p[row * 1024 + tid + i * 256] = (_Float16)(v[i] * inv);
}

// ---- final transpose: out[b][0][c][t] = xf[b][t][c] ------------------------
__global__ void k_outT(const float* __restrict__ xf, float* __restrict__ out) {
  __shared__ float tile[32][33];
  int b = blockIdx.z;
  int t0 = blockIdx.x * 32, c0 = blockIdx.y * 32;
  int tx = threadIdx.x, ty = threadIdx.y;
  #pragma unroll
  for (int i = 0; i < 4; ++i) {
    int t = t0 + ty + i * 8;
    tile[ty + i * 8][tx] = xf[(long)b * 1048576 + (long)t * 1024 + c0 + tx];
  }
  __syncthreads();
  #pragma unroll
  for (int i = 0; i < 4; ++i) {
    int c = c0 + ty + i * 8;
    out[(long)b * 1048576 + (long)c * 1024 + t0 + tx] = tile[tx][ty + i * 8];
  }
}

// ---------------------------------------------------------------------------
extern "C" void kernel_launch(void* const* d_in, const int* in_sizes, int n_in,
                              void* d_out, int out_size, void* d_ws, size_t ws_size,
                              hipStream_t stream) {
  (void)in_sizes; (void)n_in; (void)out_size; (void)ws_size;
  const float* x    = (const float*)d_in[0];
  const float* in_w = (const float*)d_in[1];
  const float* in_b = (const float*)d_in[2];
  const float* g1   = (const float*)d_in[3];
  const float* b1   = (const float*)d_in[4];
  const float* Wq   = (const float*)d_in[5];
  const float* qc   = (const float*)d_in[6];
  const float* Wk   = (const float*)d_in[7];
  const float* kc   = (const float*)d_in[8];
  const float* Wv   = (const float*)d_in[9];
  const float* vc   = (const float*)d_in[10];
  const float* Wo   = (const float*)d_in[11];
  const float* relw = (const float*)d_in[12];
  const float* g2   = (const float*)d_in[13];
  const float* b2   = (const float*)d_in[14];
  const float* W1   = (const float*)d_in[15];
  const float* W2   = (const float*)d_in[16];

  float* out = (float*)d_out;                  // [4,1,1024,1024]
  float* qk  = out + 4L * 1048576;             // [4,4,1024,1024] pre-softmax

  const long MB = 1024 * 1024;
  char* ws = (char*)d_ws;
  float*    xf   = (float*)(ws);               // 16 MiB (xf, later xf3)
  _Float16* z    = (_Float16*)(ws + 16 * MB);  // 8 MiB (z1 / z2)
  _Float16* qh   = (_Float16*)(ws + 24 * MB);
  _Float16* kh   = (_Float16*)(ws + 32 * MB);
  _Float16* vh   = (_Float16*)(ws + 40 * MB);
  _Float16* qp   = (_Float16*)(ws + 48 * MB);
  _Float16* kp   = (_Float16*)(ws + 56 * MB);
  _Float16* vp   = (_Float16*)(ws + 64 * MB);
  _Float16* prob = (_Float16*)(ws + 48 * MB);  // 32 MiB (reuses pre bufs)
  _Float16* hbuf = (_Float16*)(ws + 48 * MB);  // 32 MiB (reuses prob)
  _Float16* abuf = (_Float16*)(ws + 80 * MB);  // 8 MiB
  float*    xf2  = (float*)(ws + 88 * MB);     // 16 MiB
  _Float16* vT   = (_Float16*)(ws + 104 * MB); // 8 MiB: v^T per batch [b][c][t]
  _Float16* relT = (_Float16*)(ws + 112 * MB); // 0.5 MiB: rel_w^T f16 [1024][256]

  typedef StageNT_H<1024> Ah1k;                // f16 activations, ld=1024
  typedef StageNT_H<4096> Ah4k;                // f16 activations, ld=4096
  typedef StageNT_F<1024> Wf1k;                // f32 weights, ld=1024 (BT direct)
  typedef StageNT_F<4096> Wf4k;                // f32 weights, ld=4096

  dim3 tb32(32, 8);
  // 1) in-projection + transpose
  k_inproj<<<dim3(32, 32, 4), tb32, 0, stream>>>(x, in_w, in_b, xf);
  // 2) LN1 -> z (f16); rel_w^T (independent, overlaps)
  k_layernorm<<<4096, 256, 0, stream>>>(xf, g1, b1, z);
  k_transpose_rel<<<dim3(32, 8), tb32, 0, stream>>>(relw, relT);
  // 3) Q/K/V projections: [4096,1024] x W^T (BT = W directly)
  k_gemm<Ah1k, Wf1k, 0, 1024, 1024><<<dim3(8, 32), 256, 0, stream>>>(z, Wq, qp, nullptr);
  k_gemm<Ah1k, Wf1k, 0, 1024, 1024><<<dim3(8, 32), 256, 0, stream>>>(z, Wk, kp, nullptr);
  k_gemm<Ah1k, Wf1k, 0, 1024, 1024><<<dim3(8, 32), 256, 0, stream>>>(z, Wv, vp, nullptr);
  // 4) depthwise convs along t; then v^T for the prob@v GEMM
  k_dwconv<<<16384, 256, 0, stream>>>(qp, qc, qh);
  k_dwconv<<<16384, 256, 0, stream>>>(kp, kc, kh);
  k_dwconv<<<16384, 256, 0, stream>>>(vp, vc, vh);
  k_transpose_h<<<dim3(32, 32, 4), tb32, 0, stream>>>(vh, vT);
  // 5) attention scores per (batch, band) -> qk in d_out (pre-softmax output)
  for (int bn = 0; bn < 16; ++bn) {
    int b = bn >> 2, band = bn & 3;
    const _Float16* qs = qh + (long)b * 1048576 + band * 256;
    const _Float16* ks = kh + (long)b * 1048576 + band * 256;
    k_scores<<<dim3(8, 8), 256, 0, stream>>>(qs, ks, relT, qk + (long)bn * 1048576);
  }
  // 6) softmax over t2
  k_softmax<<<16384, 256, 0, stream>>>(qk, prob);
  // 7) a = prob @ v : BT = vT band slice [d][t2] direct, N=256 per band
  for (int bn = 0; bn < 16; ++bn) {
    int b = bn >> 2, band = bn & 3;
    const _Float16* vs = vT + (long)b * 1048576 + (long)band * 256 * 1024;
    _Float16* as = abuf + (long)b * 1048576 + band * 256;
    k_gemm<Ah1k, Ah1k, 0, 1024, 1024><<<dim3(2, 8), 256, 0, stream>>>(
        prob + (long)bn * 1048576, vs, as, nullptr);
  }
  // 8) out-projection + residual: xf2 = xf + a @ Wo^T
  k_gemm<Ah1k, Wf1k, 2, 1024, 1024><<<dim3(8, 32), 256, 0, stream>>>(abuf, Wo, xf2, xf);
  // 9) LN2 -> z (f16, reuse)
  k_layernorm<<<4096, 256, 0, stream>>>(xf2, g2, b2, z);
  // 10) FFN1: h = relu(z @ W1^T)^2   -> [4096,4096] f16
  k_gemm<Ah1k, Wf1k, 1, 4096, 1024><<<dim3(32, 32), 256, 0, stream>>>(z, W1, hbuf, nullptr);
  // 11) FFN2 + residual: xf3 = xf2 + h @ W2^T  (into xf buffer, now dead)
  k_gemm<Ah4k, Wf4k, 2, 1024, 4096><<<dim3(8, 32), 256, 0, stream>>>(hbuf, W2, xf, xf2);
  // 12) final transpose -> out
  k_outT<<<dim3(32, 32, 4), tb32, 0, stream>>>(xf, out);
}
